// SelectiveSSM_58265526337931
// MI455X (gfx1250) — matrix operand order repaired and checked by reading, not compiled, not run on hardware
//
#include <hip/hip_runtime.h>
#include <math.h>

// ---------------------------------------------------------------------------
// Selective SSM (Mamba-style) for MI455X / gfx1250, wave32 + WMMA bf16.
//   B=2, S=2048, D=1024, N=16, R=64
// Pipeline:
//   k_cvt_bf16     : x  -> bf16            (4096x1024)
//   k_pack_wproj   : [W_dtw;W_B;W_C] -> bf16 Wproj [96][1024]
//   k_cvt_bf16     : W_out -> bf16
//   k_proj_gemm    : P[4096x96] = x_bf16 @ Wproj^T              (WMMA bf16)
//   k_delta_gemm   : delta = softplus(dt @ W_dt^T + b_dt)       (WMMA bf16)
//   k_scan         : sequential SSM scan, + D*x residual folded in
//   k_layernorm    : layernorm over D, emit bf16
//   k_out_gemm     : out = ynorm @ W_out^T  (WMMA bf16; A tile staged with
//                    global_load_async_to_lds_b128/ASYNCcnt, B tile staged
//                    with one TDM tensor_load_to_lds descriptor/TENSORcnt,
//                    double buffered)
// ---------------------------------------------------------------------------

typedef __attribute__((ext_vector_type(16))) __bf16   v16bf;
typedef __attribute__((ext_vector_type(8)))  float    v8f;
typedef __attribute__((ext_vector_type(4)))  unsigned u32x4;
typedef __attribute__((ext_vector_type(8)))  unsigned u32x8;

#define BB   2
#define SS   2048
#define DD   1024
#define NN   16
#define RR   64
#define MTOT (BB * SS)      // 4096 rows
#define PCOLS 96            // dt(64) | Bm(16) | Cm(16)

// ---------------- WMMA fragment loaders (contiguous k, same in A and B) -----------
// A: 16x32 bf16, row m = lane&15, half = lane>>4.
//    elements 0..15 <-> K = 16*half + i
__device__ __forceinline__ void load_a_bf16(v16bf& a, const __bf16* base,
                                            int ld, int m, int half, int k0) {
    const __bf16* p = base + (size_t)m * ld + k0;
#pragma unroll
    for (int i = 0; i < 8; ++i) a[i] = p[16 * half + i];
#pragma unroll
    for (int i = 0; i < 8; ++i) a[8 + i] = p[16 * half + 8 + i];
}

__device__ __forceinline__ void load_a_f32cvt(v16bf& a, const float* __restrict__ base,
                                              int ld, int m, int half, int k0) {
    const float* p = base + (size_t)m * ld + k0;
#pragma unroll
    for (int i = 0; i < 8; ++i) a[i] = (__bf16)p[16 * half + i];
#pragma unroll
    for (int i = 0; i < 8; ++i) a[8 + i] = (__bf16)p[16 * half + 8 + i];
}

// B: 32x16 bf16, weights stored row-major [Nrows][K] (row n contiguous in k).
//    lane col n = lane&15; elements 0..15 <-> K = 16*half + i
__device__ __forceinline__ void load_b_bf16(v16bf& b, const __bf16* base,
                                            int ld, int n, int half, int k0) {
    const __bf16* p = base + (size_t)n * ld + k0 + 16 * half;
#pragma unroll
    for (int i = 0; i < 16; ++i) b[i] = p[i];
}

__device__ __forceinline__ void load_b_f32cvt(v16bf& b, const float* __restrict__ base,
                                              int ld, int n, int half, int k0) {
    const float* p = base + (size_t)n * ld + k0 + 16 * half;
#pragma unroll
    for (int i = 0; i < 16; ++i) b[i] = (__bf16)p[i];
}

// ---------------- gfx1250 async global->LDS copy (ASYNCcnt) ----------------
__device__ __forceinline__ void async_copy_b128(unsigned lds_off, const void* gaddr) {
    asm volatile("global_load_async_to_lds_b128 %0, %1, off"
                 :: "v"(lds_off), "v"(gaddr)
                 : "memory");
}
__device__ __forceinline__ void wait_asynccnt0() {
    asm volatile("s_wait_asynccnt 0" ::: "memory");
}

// ---------------- gfx1250 Tensor Data Mover (TENSORcnt) --------------------
// One 2D-tile DMA: LDS[lds_addr + r*tile_d0 + x] = T[r][x] for the tile at
// gaddr inside a row-major tensor with dim0 stride `stride0` (elements).
// D# per ISA 8.3/8.4: group0 = {count=1 | lds_addr | global_addr | type=2},
// group1 = {data_size=2B, tensor dims, tile dims, dim0 stride}.
__device__ __forceinline__ void tdm_load_2d_bf16(unsigned lds_addr, const void* gaddr,
                                                 unsigned tensor_d0, unsigned tensor_d1,
                                                 unsigned tile_d0, unsigned tile_d1,
                                                 unsigned stride0) {
    unsigned long long ga = (unsigned long long)gaddr;
    u32x4 g0;
    g0[0] = 1u;                                            // count=1, user D#
    g0[1] = lds_addr;                                      // LDS byte address
    g0[2] = (unsigned)(ga & 0xffffffffu);                  // global_addr[31:0]
    g0[3] = (unsigned)((ga >> 32) & 0x1ffffffu)            // global_addr[56:32]
          | (2u << 30);                                    // type=2 ("image")
    u32x8 g1;
    g1[0] = (1u << 16);                                    // data_size=1 (2 bytes)
    g1[1] = (tensor_d0 & 0xffffu) << 16;                   // tensor_dim0[15:0]
    g1[2] = (tensor_d0 >> 16) | ((tensor_d1 & 0xffffu) << 16);
    g1[3] = (tensor_d1 >> 16) | (tile_d0 << 16);           // tile_dim0
    g1[4] = tile_d1 & 0xffffu;                             // tile_dim1 (tile_dim2=0)
    g1[5] = stride0;                                       // tensor_dim0_stride[31:0]
    g1[6] = 0;                                             // stride0[47:32], stride1[15:0]
    g1[7] = 0;                                             // stride1[47:16]
    asm volatile("tensor_load_to_lds %0, %1"
                 :: "s"(g0), "s"(g1)
                 : "memory");
}

// ---------------- conversion / packing kernels -----------------------------
__global__ void k_cvt_bf16(const float* __restrict__ in, __bf16* __restrict__ out, int n) {
    int i = blockIdx.x * blockDim.x + threadIdx.x;
    int stride = gridDim.x * blockDim.x;
    for (; i < n; i += stride) out[i] = (__bf16)in[i];
}

__global__ void k_pack_wproj(const float* __restrict__ W_dtw,   // [64][1024]
                             const float* __restrict__ W_B,     // [16][1024]
                             const float* __restrict__ W_C,     // [16][1024]
                             __bf16* __restrict__ wp) {         // [96][1024]
    int i = blockIdx.x * blockDim.x + threadIdx.x;   // 96*1024 total
    if (i >= PCOLS * DD) return;
    int r = i / DD, d = i - r * DD;
    float v;
    if (r < 64)      v = W_dtw[r * DD + d];
    else if (r < 80) v = W_B[(r - 64) * DD + d];
    else             v = W_C[(r - 80) * DD + d];
    wp[i] = (__bf16)v;
}

// ---------------- projection GEMM: P = x_bf16 @ Wproj^T --------------------
// grid.x = 256 (m-tiles), block = 192 (6 waves, one per 16-col n-tile)
__global__ void k_proj_gemm(const __bf16* __restrict__ xb,   // [4096][1024]
                            const __bf16* __restrict__ wp,   // [96][1024]
                            float* __restrict__ P) {         // [4096][96]
    const int lane = threadIdx.x & 31;
    const int wave = threadIdx.x >> 5;      // n-tile 0..5
    const int half = lane >> 4, lm = lane & 15;
    const int mtile = blockIdx.x;

    const __bf16* abase = xb + (size_t)mtile * 16 * DD;
    const __bf16* bbase = wp + (size_t)wave * 16 * DD;

    v8f acc = {};
    for (int k0 = 0; k0 < DD; k0 += 32) {
        if (k0 + 32 < DD) {  // global_prefetch_b8 hint for next K-chunk
            __builtin_prefetch(abase + (size_t)lm * DD + k0 + 32, 0, 1);
            __builtin_prefetch(bbase + (size_t)lm * DD + k0 + 32, 0, 1);
        }
        v16bf a, b;
        load_a_bf16(a, abase, DD, lm, half, k0);
        load_b_bf16(b, bbase, DD, lm, half, k0);
        acc = __builtin_amdgcn_wmma_f32_16x16x32_bf16(false, a, false, b,
                                                      (short)0, acc, false, false);
    }
    float* outp = P + (size_t)mtile * 16 * PCOLS + wave * 16 + lm;
#pragma unroll
    for (int r = 0; r < 8; ++r) outp[(size_t)(r + 8 * half) * PCOLS] = acc[r];
}

// ---------------- delta GEMM: delta = softplus(dt @ W_dt^T + b_dt) ---------
// grid = (256, 8), block = 256 (8 waves); ntile = by*8 + wave (0..63)
__global__ void k_delta_gemm(const float* __restrict__ P,     // [4096][96], cols 0..63 = dt
                             const float* __restrict__ Wdt,   // [1024][64]
                             const float* __restrict__ bdt,   // [1024]
                             float* __restrict__ delta) {     // [4096][1024]
    const int lane = threadIdx.x & 31;
    const int wave = threadIdx.x >> 5;
    const int half = lane >> 4, lm = lane & 15;
    const int mtile = blockIdx.x;
    const int ntile = blockIdx.y * 8 + wave;

    v8f acc = {};
#pragma unroll
    for (int k0 = 0; k0 < RR; k0 += 32) {
        v16bf a, b;
        load_a_f32cvt(a, P + (size_t)mtile * 16 * PCOLS, PCOLS, lm, half, k0);
        load_b_f32cvt(b, Wdt + (size_t)ntile * 16 * RR, RR, lm, half, k0);
        acc = __builtin_amdgcn_wmma_f32_16x16x32_bf16(false, a, false, b,
                                                      (short)0, acc, false, false);
    }
    const int dcol = ntile * 16 + lm;
    const float bias = bdt[dcol];
#pragma unroll
    for (int r = 0; r < 8; ++r) {
        int m = mtile * 16 + r + 8 * half;
        float z = acc[r] + bias;
        float sp = (z > 20.f) ? z : log1pf(__expf(z));   // softplus
        delta[(size_t)m * DD + dcol] = sp;
    }
}

// ---------------- sequential scan ------------------------------------------
// grid = 8 blocks x 256 threads; block bi: b = bi>>2, d = (bi&3)*256 + tid.
// Each thread owns one (b,d) channel with NN=16 states in registers.
// Bm/Cm staged in LDS, 64 steps per chunk (8 KB).
#define SCAN_CH 64
__global__ void k_scan(const float* __restrict__ delta,  // [4096][1024]
                       const float* __restrict__ x,      // [4096][1024]
                       const float* __restrict__ P,      // [4096][96] (64..95 = Bm|Cm)
                       const float* __restrict__ A_log,  // [1024][16]
                       const float* __restrict__ Dp,     // [1024]
                       float* __restrict__ ysc) {        // [4096][1024]
    __shared__ float sBC[SCAN_CH][32];                   // [step][0..15 B | 16..31 C]
    const int tid = threadIdx.x;
    const int bi  = blockIdx.x;
    const int b   = bi >> 2;
    const int d   = (bi & 3) * 256 + tid;

    float An[NN];
#pragma unroll
    for (int n = 0; n < NN; ++n) An[n] = -__expf(A_log[d * NN + n]);
    float h[NN];
#pragma unroll
    for (int n = 0; n < NN; ++n) h[n] = 0.f;
    const float dpv = Dp[d];

    for (int s0 = 0; s0 < SS; s0 += SCAN_CH) {
        for (int t = tid; t < SCAN_CH * 32; t += 256) {
            int step = t >> 5, j = t & 31;
            sBC[step][j] = P[(size_t)(b * SS + s0 + step) * PCOLS + 64 + j];
        }
        __syncthreads();
#pragma unroll 4
        for (int step = 0; step < SCAN_CH; ++step) {
            const size_t m = (size_t)(b * SS + s0 + step);
            const float dl = delta[m * DD + d];
            const float xv = x[m * DD + d];
            const float dx = dl * xv;
            float y = 0.f;
#pragma unroll
            for (int n = 0; n < NN; ++n) {
                float dA = __expf(dl * An[n]);
                h[n] = dA * h[n] + dx * sBC[step][n];
                y += h[n] * sBC[step][16 + n];
            }
            ysc[m * DD + d] = y + dpv * xv;              // fold D*x residual
        }
        __syncthreads();
    }
}

// ---------------- layernorm over D, emit bf16 ------------------------------
__global__ void k_layernorm(const float* __restrict__ y,   // [4096][1024]
                            const float* __restrict__ g,
                            const float* __restrict__ bta,
                            __bf16* __restrict__ outb) {   // [4096][1024]
    __shared__ float s1[256], s2[256];
    const int m = blockIdx.x, tid = threadIdx.x;
    const float* row = y + (size_t)m * DD;
    float sum = 0.f, ss = 0.f;
    for (int i = tid; i < DD; i += 256) { float v = row[i]; sum += v; ss += v * v; }
    s1[tid] = sum; s2[tid] = ss;
    __syncthreads();
    for (int o = 128; o > 0; o >>= 1) {
        if (tid < o) { s1[tid] += s1[tid + o]; s2[tid] += s2[tid + o]; }
        __syncthreads();
    }
    const float mu = s1[0] * (1.f / DD);
    const float var = s2[0] * (1.f / DD) - mu * mu;
    const float rs = rsqrtf(var + 1e-5f);
    for (int i = tid; i < DD; i += 256) {
        float v = (row[i] - mu) * rs * g[i] + bta[i];
        outb[(size_t)m * DD + i] = (__bf16)v;
    }
}

// ---------------- output GEMM: out = ynorm @ W_out^T -----------------------
// grid = (256, 8), block = 256 (8 waves).  Double-buffered LDS staging:
//   A tile  16 x 64 bf16 =  2 KB : per-lane global_load_async_to_lds_b128
//   B tile 128 x 64 bf16 = 16 KB : ONE tensor_load_to_lds TDM descriptor
// Chunk pipeline: s_wait_asynccnt 0 (+ wave0 s_wait_tensorcnt 0) -> barrier
// -> issue next buffer's fills -> 2 WMMA steps from LDS.
#define KC 64
__global__ void k_out_gemm(const __bf16* __restrict__ yb,   // [4096][1024]
                           const __bf16* __restrict__ wo,   // [1024][1024]
                           float* __restrict__ out) {       // [4096][1024]
    __shared__ __align__(16) __bf16 sA[2][16 * KC];    //  2 x 2 KB
    __shared__ __align__(16) __bf16 sB[2][128 * KC];   //  2 x 16 KB

    const int tid  = threadIdx.x;
    const int lane = tid & 31;
    const int wave = tid >> 5;            // 0..7 -> n-tile within block
    const int half = lane >> 4, lm = lane & 15;
    const int mtile = blockIdx.x;
    const int nbase = blockIdx.y * 8;     // first n-tile of this block

    // issue the fills of LDS buffer `buf` with K-chunk starting at k0
    auto stage = [&](int k0, int buf) {
        // A tile: 16 rows x 8 x 16B, one async b128 per thread (tid < 128)
        if (tid < 128) {
            int r = tid >> 3, c = tid & 7;
            async_copy_b128((unsigned)(size_t)&sA[buf][r * KC + c * 8],
                            yb + (size_t)(mtile * 16 + r) * DD + k0 + c * 8);
        }
        // B tile: one TDM 2D-tile DMA (EXEC-independent, wave 0 issues)
        if (wave == 0) {
            tdm_load_2d_bf16((unsigned)(size_t)&sB[buf][0],
                             wo + (size_t)(nbase * 16) * DD + k0,
                             /*tensor_d0=*/DD, /*tensor_d1=*/DD,
                             /*tile_d0=*/KC, /*tile_d1=*/128,
                             /*stride0=*/DD);
        }
    };

    stage(0, 0);

    v8f acc = {};
    const int nch = DD / KC;                     // 16 chunks
    for (int ch = 0; ch < nch; ++ch) {
        wait_asynccnt0();                        // our A-tile fills landed
        if (wave == 0) __builtin_amdgcn_s_wait_tensorcnt(0);  // B tile landed
        __syncthreads();                         // everyone sees both tiles
        if (ch + 1 < nch) stage((ch + 1) * KC, (ch + 1) & 1);

        const __bf16* sa = sA[ch & 1];
        const __bf16* sb = sB[ch & 1] + (size_t)wave * 16 * KC;
#pragma unroll
        for (int kk = 0; kk < KC; kk += 32) {
            v16bf a, b;
            load_a_bf16(a, sa, KC, lm, half, kk);     // ds_load fragments
            load_b_bf16(b, sb, KC, lm, half, kk);
            acc = __builtin_amdgcn_wmma_f32_16x16x32_bf16(false, a, false, b,
                                                          (short)0, acc, false, false);
        }
    }

    const int e = (nbase + wave) * 16 + lm;
#pragma unroll
    for (int r = 0; r < 8; ++r)
        out[(size_t)(mtile * 16 + r + 8 * half) * DD + e] = acc[r];
}

// ---------------------------------------------------------------------------
extern "C" void kernel_launch(void* const* d_in, const int* in_sizes, int n_in,
                              void* d_out, int out_size, void* d_ws, size_t ws_size,
                              hipStream_t stream) {
    (void)in_sizes; (void)n_in; (void)out_size; (void)ws_size;
    const float* x     = (const float*)d_in[0];
    const float* W_dtw = (const float*)d_in[1];
    const float* W_dt  = (const float*)d_in[2];
    const float* b_dt  = (const float*)d_in[3];
    const float* A_log = (const float*)d_in[4];
    const float* W_B   = (const float*)d_in[5];
    const float* W_C   = (const float*)d_in[6];
    const float* Dp    = (const float*)d_in[7];
    const float* ln_g  = (const float*)d_in[8];
    const float* ln_b  = (const float*)d_in[9];
    const float* W_out = (const float*)d_in[10];
    float* out = (float*)d_out;

    // workspace layout (256B aligned chunks)
    char* ws = (char*)d_ws;
    size_t off = 0;
    auto take = [&](size_t bytes) -> void* {
        void* p = (void*)(ws + off);
        off += (bytes + 255) & ~(size_t)255;
        return p;
    };
    float*  P     = (float*)take((size_t)MTOT * PCOLS * 4);   // 1.5 MB
    float*  delta = (float*)take((size_t)MTOT * DD * 4);      // 16 MB
    float*  ysc   = (float*)take((size_t)MTOT * DD * 4);      // 16 MB
    __bf16* xb    = (__bf16*)take((size_t)MTOT * DD * 2);     // 8 MB
    __bf16* wproj = (__bf16*)take((size_t)PCOLS * DD * 2);    // 192 KB
    __bf16* wob   = (__bf16*)take((size_t)DD * DD * 2);       // 2 MB
    __bf16* ynb   = (__bf16*)take((size_t)MTOT * DD * 2);     // 8 MB

    // 0) precision conversions / weight packing
    k_cvt_bf16<<<dim3((MTOT * DD + 255) / 256), dim3(256), 0, stream>>>(x, xb, MTOT * DD);
    k_pack_wproj<<<dim3((PCOLS * DD + 255) / 256), dim3(256), 0, stream>>>(W_dtw, W_B, W_C, wproj);
    k_cvt_bf16<<<dim3((DD * DD + 255) / 256), dim3(256), 0, stream>>>(W_out, wob, DD * DD);

    // 1) fused projections: P = x @ [W_dtw;W_B;W_C]^T
    k_proj_gemm<<<dim3(MTOT / 16), dim3(192), 0, stream>>>(xb, wproj, P);

    // 2) delta = softplus(dt @ W_dt^T + b_dt)
    k_delta_gemm<<<dim3(MTOT / 16, 8), dim3(256), 0, stream>>>(P, W_dt, b_dt, delta);

    // 3) sequential selective scan (+ D*x residual)
    k_scan<<<dim3(8), dim3(256), 0, stream>>>(delta, x, P, A_log, Dp, ysc);

    // 4) layernorm -> bf16
    k_layernorm<<<dim3(MTOT), dim3(256), 0, stream>>>(ysc, ln_g, ln_b, ynb);

    // 5) out = ynorm @ W_out^T (async + TDM staged WMMA)
    k_out_gemm<<<dim3(MTOT / 16, 8), dim3(256), 0, stream>>>(ynb, wob, out);
}
